// Circulant_layer_28595892256901
// MI455X (gfx1250) — compile-verified, matrix operand order checked by reading, not run on hardware
//
#include <hip/hip_runtime.h>

typedef __attribute__((ext_vector_type(16))) __bf16 v16bf;
typedef __attribute__((ext_vector_type(8)))  float  v8f;
typedef __attribute__((ext_vector_type(4)))  float  v4f;
typedef __attribute__((ext_vector_type(4)))  int    v4i;

#define D_IN  1024
#define N_OUT 1024
#define BATCH 128

#if __has_builtin(__builtin_amdgcn_global_load_async_to_lds_b128)
#define HAVE_ASYNC_LDS 1
#else
#define HAVE_ASYNC_LDS 0
#endif

static __device__ __forceinline__ void wait_asynccnt0() {
#if HAVE_ASYNC_LDS
#if __has_builtin(__builtin_amdgcn_s_wait_asynccnt)
    __builtin_amdgcn_s_wait_asynccnt(0);
#else
    asm volatile("s_wait_asynccnt 0" ::: "memory");
#endif
#endif
}

// ---------------------------------------------------------------------------
// Stage 1: dmat = des @ kern, bmat = body @ kern (fused).
// Block = 4 waves = 128 threads covering 64 M-rows x 16 N-cols.
// Per K-step of 32, the 32x16 f32 kern tile (2KB) is staged into LDS with one
// GLOBAL_LOAD_ASYNC_TO_LDS_B128 (128 lanes x 16B), tracked by ASYNCcnt.
// All 4 waves share the staged tile for their B-fragment; each wave runs two
// v_wmma_f32_16x16x32_bf16 (des path + body path) per K-step.
// ---------------------------------------------------------------------------
__global__ __launch_bounds__(128) void gemm_dual_bf16(
    const float* __restrict__ des, const float* __restrict__ body,
    const float* __restrict__ kern, float* __restrict__ dmat,
    float* __restrict__ bmat)
{
    __shared__ float ktile[32 * 16];   // K-chunk x N-tile, row-major

    const int t    = threadIdx.x;
    const int lane = t & 31;
    const int wave = t >> 5;           // 0..3 -> M sub-tile
    const int half = lane >> 4;        // K-subgroup selector
    const int r    = lane & 15;

    const int n0   = blockIdx.x * 16;                  // 64 N-tiles
    const int mRow = blockIdx.y * 64 + wave * 16 + r;  // 2 M-groups of 64

    const float* __restrict__ rowD = des  + mRow * D_IN;
    const float* __restrict__ rowB = body + mRow * D_IN;

    // Staging assignment: thread t moves 16B: row sRow (0..31), cols sCol..sCol+3
    const int sRow = t >> 2;
    const int sCol = (t & 3) * 4;
    const float* __restrict__ gK = kern + sRow * N_OUT + n0 + sCol;
    float* lK = &ktile[sRow * 16 + sCol];

    v8f accD = {};
    v8f accB = {};

    for (int kk = 0; kk < D_IN; kk += 32) {
#if HAVE_ASYNC_LDS
        __builtin_amdgcn_global_load_async_to_lds_b128(
            (v4i*)(gK + kk * N_OUT), (v4i*)lK, 0, 0);
        wait_asynccnt0();
#else
        *(v4f*)lK = *(const v4f*)(gK + kk * N_OUT);
#endif
        __syncthreads();

        v16bf aD, aB, bK;
        const int kA0 = kk + half * 8;   // first run of 8 consecutive K
        const int kA1 = kA0 + 16;        // second run
#pragma unroll
        for (int e = 0; e < 8; ++e) {
            aD[e]     = (__bf16)rowD[kA0 + e];
            aD[e + 8] = (__bf16)rowD[kA1 + e];
            aB[e]     = (__bf16)rowB[kA0 + e];
            aB[e + 8] = (__bf16)rowB[kA1 + e];
            // B-fragment from the staged LDS tile (local k row, column r)
            bK[e]     = (__bf16)ktile[(half * 8 + e) * 16 + r];
            bK[e + 8] = (__bf16)ktile[(16 + half * 8 + e) * 16 + r];
        }

        accD = __builtin_amdgcn_wmma_f32_16x16x32_bf16(
            false, aD, false, bK, (short)0, accD, false, false);
        accB = __builtin_amdgcn_wmma_f32_16x16x32_bf16(
            false, aB, false, bK, (short)0, accB, false, false);

        __syncthreads();   // protect ktile before next staging round
    }

    // C/D layout: lane r -> N=r; VGPR v -> M = v + 8*half
    const int nOut = n0 + r;
#pragma unroll
    for (int v = 0; v < 8; ++v) {
        const int mOut = blockIdx.y * 64 + wave * 16 + v + 8 * half;
        dmat[mOut * N_OUT + nOut] = accD[v];
        bmat[mOut * N_OUT + nOut] = accB[v];
    }
}

// ---------------------------------------------------------------------------
// Stage 2: out[b,k] = 2 * sum_j bmat[b,(k-j)&1023] * dmat[b,j]
// (mv1 == mv2 since circular convolution is commutative.)
// One block per batch row; both 1024-float vectors staged in LDS.
// Consecutive lanes -> consecutive bs[] banks; ds[j] is a broadcast.
// ---------------------------------------------------------------------------
__global__ __launch_bounds__(256) void circ_conv_kernel(
    const float* __restrict__ dmat, const float* __restrict__ bmat,
    float* __restrict__ out)
{
    __shared__ float ds[D_IN];
    __shared__ float bs[D_IN];
    const int b   = blockIdx.x;
    const int tid = threadIdx.x;

#pragma unroll
    for (int i = 0; i < 4; ++i) {
        const int idx = tid + 256 * i;
        ds[idx] = dmat[b * D_IN + idx];
        bs[idx] = bmat[b * D_IN + idx];
    }
    __syncthreads();

    float acc0 = 0.f, acc1 = 0.f, acc2 = 0.f, acc3 = 0.f;
#pragma unroll 4
    for (int j = 0; j < D_IN; ++j) {
        const float dj = ds[j];
        const int base = (tid - j) & (D_IN - 1);
        acc0 += bs[base]                      * dj;
        acc1 += bs[(base + 256) & (D_IN - 1)] * dj;
        acc2 += bs[(base + 512) & (D_IN - 1)] * dj;
        acc3 += bs[(base + 768) & (D_IN - 1)] * dj;
    }

    out[b * D_IN + tid      ] = 2.0f * acc0;
    out[b * D_IN + tid + 256] = 2.0f * acc1;
    out[b * D_IN + tid + 512] = 2.0f * acc2;
    out[b * D_IN + tid + 768] = 2.0f * acc3;
}

// ---------------------------------------------------------------------------
extern "C" void kernel_launch(void* const* d_in, const int* in_sizes, int n_in,
                              void* d_out, int out_size, void* d_ws, size_t ws_size,
                              hipStream_t stream)
{
    (void)in_sizes; (void)n_in; (void)out_size; (void)ws_size;

    const float* des  = (const float*)d_in[0];
    const float* body = (const float*)d_in[1];
    const float* kern = (const float*)d_in[2];

    float* dmat = (float*)d_ws;                 // 128*1024 f32
    float* bmat = dmat + BATCH * D_IN;          // 128*1024 f32
    float* out  = (float*)d_out;

    gemm_dual_bf16<<<dim3(64, 2), 128, 0, stream>>>(des, body, kern, dmat, bmat);
    circ_conv_kernel<<<BATCH, 256, 0, stream>>>(dmat, bmat, out);
}
